// Model_49649821942101
// MI455X (gfx1250) — matrix operand order repaired and checked by reading, not compiled, not run on hardware
//
#include <hip/hip_runtime.h>
#include <hip/hip_bf16.h>

// ---------------- problem constants ----------------
#define BSZ   64
#define SEQ   512
#define PRED  96
#define DDIM  321
#define HID   2048
#define WIN   64
#define STRD  32
#define FT    33
#define NWIN  17
#define EMBED 1088          // NWIN*WIN
#define ROWS  (BSZ*DDIM)    // 20544 (= 1284 * 16)
#define LOWN  9
#define MIDN  13
#define HIGHN 11
#define LHN   18
#define MHN   26
#define PI_F  3.14159265358979323846f

typedef __attribute__((ext_vector_type(16))) _Float16 v16h;
typedef __attribute__((ext_vector_type(8)))  float    v8f;
typedef unsigned int uint;

__device__ __forceinline__ float lrelu(float v) { return v > 0.f ? v : 0.01f * v; }

// A-matrix 16x32 f16 fragment: lane holds row M=lane&15.
// lanes 0-15: K = {0..7, 16..23}; lanes 16-31: K = {8..15, 24..31}
// -> two contiguous 16B chunks at half-offsets 8*hi and 16+8*hi.
__device__ __forceinline__ v16h load_frag_a(const _Float16* rowbase, int hi) {
    union { v16h h; int4 q[2]; } f;
    f.q[0] = *(const int4*)(rowbase + 8 * hi);
    f.q[1] = *(const int4*)(rowbase + 16 + 8 * hi);
    return f.h;
}
// B-matrix 32x16 f16 fragment: lane holds column N=lane&15.
// lanes 0-15: K = {0..7, 16..23}; lanes 16-31: K = {8..15, 24..31} (same as A).
__device__ __forceinline__ v16h load_frag_b(const _Float16* colbase, int hi) {
    union { v16h h; int4 q[2]; } f;
    f.q[0] = *(const int4*)(colbase + 8 * hi);
    f.q[1] = *(const int4*)(colbase + 16 + 8 * hi);
    return f.h;
}

// ---------------- RevIN stats: coalesced (lanes sweep d) ----------------
#define DCHUNKS 6   // ceil(321/64)
__global__ __launch_bounds__(256) void k_stats(const float* __restrict__ x,
                                               float* __restrict__ mean,
                                               float* __restrict__ stdv) {
    __shared__ float s1[256], s2[256];
    int dl = threadIdx.x & 63;
    int tg = threadIdx.x >> 6;           // 0..3
    int dchunk = blockIdx.x % DCHUNKS;
    int b = blockIdx.x / DCHUNKS;
    int d = dchunk * 64 + dl;
    float a = 0.f, q = 0.f;
    if (d < DDIM) {
        for (int t = tg; t < SEQ; t += 4) {
            float v = x[(size_t)b * SEQ * DDIM + (size_t)t * DDIM + d];
            a += v; q += v * v;
        }
    }
    s1[threadIdx.x] = a; s2[threadIdx.x] = q;
    __syncthreads();
    if (tg == 0 && d < DDIM) {
        float sa = s1[dl] + s1[dl + 64] + s1[dl + 128] + s1[dl + 192];
        float sq = s2[dl] + s2[dl + 64] + s2[dl + 128] + s2[dl + 192];
        float m = sa / (float)SEQ;
        float var = sq / (float)SEQ - m * m;
        mean[b * DDIM + d] = m;
        stdv[b * DDIM + d] = sqrtf(var + 1e-5f);
    }
}

// ------- tiled transpose + f32->f16: src KxN -> dst NxK (K,N multiples of 32) -------
__global__ __launch_bounds__(256) void k_transpose_f16(const float* __restrict__ src,
                                                       _Float16* __restrict__ dst,
                                                       int K, int N) {
    __shared__ float tile[32][33];
    int tx = threadIdx.x;                // 0..31
    int ty = threadIdx.y;                // 0..7
    int n0 = blockIdx.x * 32;
    int k0 = blockIdx.y * 32;
#pragma unroll
    for (int i = 0; i < 4; ++i) {
        int k = k0 + ty + i * 8;
        tile[ty + i * 8][tx] = src[(size_t)k * N + n0 + tx];
    }
    __syncthreads();
#pragma unroll
    for (int i = 0; i < 4; ++i) {
        int n = n0 + ty + i * 8;
        dst[(size_t)n * K + k0 + tx] = (_Float16)tile[tx][ty + i * 8];
    }
}

// ---------------- band MLP weights ----------------
struct BandW {
    const float *l0wr, *l0wi, *l0br, *l0bi;
    const float *l1wr, *l1wi, *l1br, *l1bi;
    const float *l2wr, *l2wi, *l2br, *l2bi;
    const float *m0wr, *m0wi, *m0br, *m0bi;
    const float *m1wr, *m1wi, *m1br, *m1bi;
    const float *h0wr, *h0wi, *h0br, *h0bi;
};

// ---------------- spectral: one block per (b, d, window) ----------------
__global__ __launch_bounds__(64) void k_spectral(const float* __restrict__ x,
                                                 const float* __restrict__ window,
                                                 const float* __restrict__ revin_w,
                                                 const float* __restrict__ revin_b,
                                                 const float* __restrict__ mean,
                                                 const float* __restrict__ stdv,
                                                 BandW bw,
                                                 _Float16* __restrict__ emb) {
    __shared__ float sw[WIN];
    __shared__ float twc[WIN], tws[WIN];   // cos/sin(2*pi*j/64)
    __shared__ float Xr[FT], Xi[FT];
    __shared__ float Yr[FT], Yi[FT];
    __shared__ float t1r[LHN], t1i[LHN], t2r[LHN], t2i[LHN];
    __shared__ float mr[MHN], mi[MHN];
    __shared__ float red[WIN];

    int tid = threadIdx.x;
    int bi = blockIdx.x;
    int w  = bi % NWIN;
    int bd = bi / NWIN;
    int d  = bd % DDIM;
    int b  = bd / DDIM;

    // one sincos per thread builds the whole twiddle table
    {
        float sn, cn;
        __sincosf(2.0f * PI_F * (float)tid / (float)WIN, &sn, &cn);
        twc[tid] = cn; tws[tid] = sn;
    }

    // window norm
    float wn = window[tid];
    red[tid] = wn * wn;
    __syncthreads();
    for (int s = 32; s > 0; s >>= 1) {
        if (tid < s) red[tid] += red[tid + s];
        __syncthreads();
    }
    float norm = sqrtf(red[0]);

    // RevIN-normalized, zero-padded, windowed sample
    int t = w * STRD + tid - STRD;
    float smp = 0.f;
    if (t >= 0 && t < SEQ) {
        float xv = x[(size_t)b * SEQ * DDIM + (size_t)t * DDIM + d];
        smp = (xv - mean[bd]) / stdv[bd] * revin_w[d] + revin_b[d];
    }
    sw[tid] = smp * (wn / norm);
    __syncthreads();

    // rfft(64): X[k] = sum_n s[n] (cos - i sin)(2 pi k n / 64)
    if (tid < FT) {
        float ar = 0.f, ai = 0.f;
#pragma unroll 8
        for (int n = 0; n < WIN; ++n) {
            int idx = (tid * n) & (WIN - 1);
            float sv = sw[n];
            ar += sv * twc[idx];
            ai -= sv * tws[idx];
        }
        Xr[tid] = ar; Xi[tid] = ai;
    }
    __syncthreads();

    // stage 1: low0 (9->18,+act), mid0 (13->26,+act), high0 (11->11 linear)
    if (tid < LHN) {
        float ar = bw.l0br[tid], ai = bw.l0bi[tid];
        for (int i = 0; i < LOWN; ++i) {
            float zr = Xr[i], zi = Xi[i];
            float wr = bw.l0wr[i * LHN + tid], wi = bw.l0wi[i * LHN + tid];
            ar += zr * wr - zi * wi;
            ai += zr * wi + zi * wr;
        }
        t1r[tid] = lrelu(ar); t1i[tid] = lrelu(ai);
    } else if (tid < LHN + MHN) {
        int j = tid - LHN;
        float ar = bw.m0br[j], ai = bw.m0bi[j];
        for (int i = 0; i < MIDN; ++i) {
            float zr = Xr[LOWN + i], zi = Xi[LOWN + i];
            float wr = bw.m0wr[i * MHN + j], wi = bw.m0wi[i * MHN + j];
            ar += zr * wr - zi * wi;
            ai += zr * wi + zi * wr;
        }
        mr[j] = lrelu(ar); mi[j] = lrelu(ai);
    } else if (tid < LHN + MHN + HIGHN) {
        int j = tid - (LHN + MHN);
        float ar = bw.h0br[j], ai = bw.h0bi[j];
        for (int i = 0; i < HIGHN; ++i) {
            float zr = Xr[LOWN + MIDN + i], zi = Xi[LOWN + MIDN + i];
            float wr = bw.h0wr[i * HIGHN + j], wi = bw.h0wi[i * HIGHN + j];
            ar += zr * wr - zi * wi;
            ai += zr * wi + zi * wr;
        }
        Yr[LOWN + MIDN + j] = ar; Yi[LOWN + MIDN + j] = ai;
    }
    __syncthreads();

    // stage 2: low1 (18->18,+act), mid1 (26->13 linear, output)
    if (tid < LHN) {
        float ar = bw.l1br[tid], ai = bw.l1bi[tid];
        for (int i = 0; i < LHN; ++i) {
            float zr = t1r[i], zi = t1i[i];
            float wr = bw.l1wr[i * LHN + tid], wi = bw.l1wi[i * LHN + tid];
            ar += zr * wr - zi * wi;
            ai += zr * wi + zi * wr;
        }
        t2r[tid] = lrelu(ar); t2i[tid] = lrelu(ai);
    } else if (tid < LHN + MIDN) {
        int j = tid - LHN;
        float ar = bw.m1br[j], ai = bw.m1bi[j];
        for (int i = 0; i < MHN; ++i) {
            float zr = mr[i], zi = mi[i];
            float wr = bw.m1wr[i * MIDN + j], wi = bw.m1wi[i * MIDN + j];
            ar += zr * wr - zi * wi;
            ai += zr * wi + zi * wr;
        }
        Yr[LOWN + j] = ar; Yi[LOWN + j] = ai;
    }
    __syncthreads();

    // stage 3: low2 (18->9 linear, output)
    if (tid < LOWN) {
        float ar = bw.l2br[tid], ai = bw.l2bi[tid];
        for (int i = 0; i < LHN; ++i) {
            float zr = t2r[i], zi = t2i[i];
            float wr = bw.l2wr[i * LOWN + tid], wi = bw.l2wi[i * LOWN + tid];
            ar += zr * wr - zi * wi;
            ai += zr * wi + zi * wr;
        }
        Yr[tid] = ar; Yi[tid] = ai;
    }
    __syncthreads();

    // irfft(64): x[n] = (1/64)[Y0r + (-1)^n Y32r + 2*sum_{k=1..31}(Ykr cos - Yki sin)]
    float acc = Yr[0] + ((tid & 1) ? -Yr[FT - 1] : Yr[FT - 1]);
#pragma unroll 8
    for (int k = 1; k < FT - 1; ++k) {
        int idx = (k * tid) & (WIN - 1);
        acc += 2.0f * (Yr[k] * twc[idx] - Yi[k] * tws[idx]);
    }
    acc *= (1.0f / (float)WIN);
    emb[(size_t)bd * EMBED + w * WIN + tid] = (_Float16)acc;
}

// ---------------- fc1: (ROWS x 1088) @ (1088 x 2048), bias + leaky, f16 out ----------------
// 8 waves per block; block tile M=16, N=128; double-buffered LDS, b128 staging,
// global_prefetch 2 K-tiles ahead.
__global__ __launch_bounds__(256) void k_fc1(const _Float16* __restrict__ emb,
                                             const _Float16* __restrict__ w1t, // 2048 x 1088
                                             const float* __restrict__ b1,
                                             _Float16* __restrict__ h) {
    __shared__ __align__(16) _Float16 As[2][16 * 32];
    __shared__ __align__(16) _Float16 Bs[2][128 * 32];

    int tid  = threadIdx.x;
    int lane = tid & 31;
    int wave = tid >> 5;                       // 0..7
    int hi   = (lane >> 4) & 1;
    int lr   = lane & 15;
    size_t mrow0 = (size_t)blockIdx.x * 16;    // 1284 m-tiles
    int ncol0    = blockIdx.y * 128;           // 16 n-groups

    // staging geometry: 16B chunks of 8 halves; A: 64 chunks, B: 512 chunks
    int arow = tid >> 2, acol = (tid & 3) * 8;          // A chunk (tid<64)
    int c1 = tid, c2 = tid + 256;                       // B chunks
    const _Float16* b1p = &w1t[(size_t)(ncol0 + (c1 >> 2)) * EMBED + (c1 & 3) * 8];
    const _Float16* b2p = &w1t[(size_t)(ncol0 + (c2 >> 2)) * EMBED + (c2 & 3) * 8];
    const _Float16* a1p = &emb[(mrow0 + arow) * EMBED + acol];

    int4 ra, rb0, rb1;
    auto load_tile = [&](int kb) {
        if (tid < 64)
            ra = *(const int4*)(a1p + kb);
        rb0 = *(const int4*)(b1p + kb);
        rb1 = *(const int4*)(b2p + kb);
    };
    auto store_tile = [&](int buf) {
        if (tid < 64)
            *(int4*)(&As[buf][arow * 32 + acol]) = ra;
        *(int4*)(&Bs[buf][(c1 >> 2) * 32 + (c1 & 3) * 8]) = rb0;
        *(int4*)(&Bs[buf][(c2 >> 2) * 32 + (c2 & 3) * 8]) = rb1;
    };

    load_tile(0);
    store_tile(0);
    __syncthreads();

    v8f c = {};
    for (int kt = 0; kt < EMBED / 32; ++kt) {
        int cur = kt & 1;
        if (kt + 1 < EMBED / 32) load_tile((kt + 1) * 32);   // tile k+1 -> regs
        if (kt + 2 < EMBED / 32) {                           // tile k+2 -> L0/L2
            int pf = (kt + 2) * 32;
            __builtin_prefetch(b1p + pf, 0, 3);              // global_prefetch_b8
            __builtin_prefetch(b2p + pf, 0, 3);
            if (tid < 64) __builtin_prefetch(a1p + pf, 0, 3);
        }
        v16h a  = load_frag_a(&As[cur][lr * 32], hi);
        v16h bf = load_frag_b(&Bs[cur][(wave * 16 + lr) * 32], hi);
        c = __builtin_amdgcn_wmma_f32_16x16x32_f16(false, a, false, bf,
                                                   (short)0, c, false, false);
        if (kt + 1 < EMBED / 32) store_tile(cur ^ 1);
        __syncthreads();   // writers of buf^1 done; readers of buf done
    }

    int ncol = ncol0 + wave * 16 + lr;
    float bias = b1[ncol];
#pragma unroll
    for (int r = 0; r < 8; ++r) {
        float v = c[r] + bias;
        v = v > 0.f ? v : 0.01f * v;
        size_t row = mrow0 + r + hi * 8;
        h[row * HID + ncol] = (_Float16)v;
    }
}

// ---------------- fc2: (ROWS x 2048) @ (2048 x 96) + bias + de-norm + transpose ----------------
__global__ __launch_bounds__(128) void k_fc2(const _Float16* __restrict__ h,
                                             const _Float16* __restrict__ w2t, // 96 x 2048
                                             const float* __restrict__ b2,
                                             const float* __restrict__ revin_w,
                                             const float* __restrict__ revin_b,
                                             const float* __restrict__ mean,
                                             const float* __restrict__ stdv,
                                             float* __restrict__ out) {
    int tid  = threadIdx.x;
    int lane = tid & 31;
    int wave = tid >> 5;
    int hi   = lane >> 4;
    int lr   = lane & 15;
    int tile  = blockIdx.x * 4 + wave;   // 1284*6 = 7704 tiles, 1926 blocks
    int mtile = tile / 6;
    int ntile = tile % 6;
    size_t arow = (size_t)(mtile * 16 + lr) * HID;
    size_t brow = (size_t)(ntile * 16 + lr) * HID;

    v8f c = {};
#pragma unroll 4
    for (int kt = 0; kt < HID / 32; ++kt) {
        int kb = kt * 32;
        if (kt + 8 < HID / 32)                       // stream h through L2->L0
            __builtin_prefetch(h + arow + kb + 256, 0, 3);
        v16h a = load_frag_a(h + arow + kb, hi);     // 2x global_load_b128
        v16h b = load_frag_b(w2t + brow + kb, hi);   // 2x global_load_b128
        c = __builtin_amdgcn_wmma_f32_16x16x32_f16(false, a, false, b,
                                                   (short)0, c, false, false);
    }
    int p = ntile * 16 + lr;
    float bias = b2[p];
#pragma unroll
    for (int r = 0; r < 8; ++r) {
        int row = mtile * 16 + r + hi * 8;   // = b*DDIM + d
        int bb = row / DDIM;
        int d  = row - bb * DDIM;
        float y = c[r] + bias;
        y = (y - revin_b[d]) / (revin_w[d] + 1e-10f) * stdv[row] + mean[row];
        out[(size_t)bb * PRED * DDIM + (size_t)p * DDIM + d] = y;
    }
}

// ---------------- host launcher ----------------
extern "C" void kernel_launch(void* const* d_in, const int* in_sizes, int n_in,
                              void* d_out, int out_size, void* d_ws, size_t ws_size,
                              hipStream_t stream) {
    const float* x       = (const float*)d_in[0];
    const float* window  = (const float*)d_in[4];
    const float* revin_w = (const float*)d_in[5];
    const float* revin_b = (const float*)d_in[6];
    BandW bw;
    bw.l0wr = (const float*)d_in[7];  bw.l0wi = (const float*)d_in[8];
    bw.l0br = (const float*)d_in[9];  bw.l0bi = (const float*)d_in[10];
    bw.l1wr = (const float*)d_in[11]; bw.l1wi = (const float*)d_in[12];
    bw.l1br = (const float*)d_in[13]; bw.l1bi = (const float*)d_in[14];
    bw.l2wr = (const float*)d_in[15]; bw.l2wi = (const float*)d_in[16];
    bw.l2br = (const float*)d_in[17]; bw.l2bi = (const float*)d_in[18];
    bw.m0wr = (const float*)d_in[19]; bw.m0wi = (const float*)d_in[20];
    bw.m0br = (const float*)d_in[21]; bw.m0bi = (const float*)d_in[22];
    bw.m1wr = (const float*)d_in[23]; bw.m1wi = (const float*)d_in[24];
    bw.m1br = (const float*)d_in[25]; bw.m1bi = (const float*)d_in[26];
    bw.h0wr = (const float*)d_in[27]; bw.h0wi = (const float*)d_in[28];
    bw.h0br = (const float*)d_in[29]; bw.h0bi = (const float*)d_in[30];
    const float* fc1_w = (const float*)d_in[31];
    const float* fc1_b = (const float*)d_in[32];
    const float* fc2_w = (const float*)d_in[33];
    const float* fc2_b = (const float*)d_in[34];

    char* ws = (char*)d_ws;
    size_t off = 0;
    auto take = [&](size_t bytes) -> char* {
        char* p = ws + off;
        off = (off + bytes + 255) & ~(size_t)255;
        return p;
    };
    float*    mean = (float*)take((size_t)ROWS * 4);
    float*    stdv = (float*)take((size_t)ROWS * 4);
    _Float16* emb  = (_Float16*)take((size_t)ROWS * EMBED * 2);
    _Float16* w1t  = (_Float16*)take((size_t)HID * EMBED * 2);
    _Float16* w2t  = (_Float16*)take((size_t)PRED * HID * 2);
    _Float16* hbuf = (_Float16*)take((size_t)ROWS * HID * 2);

    k_stats<<<BSZ * DCHUNKS, 256, 0, stream>>>(x, mean, stdv);
    k_transpose_f16<<<dim3(HID / 32, EMBED / 32), dim3(32, 8), 0, stream>>>(
        fc1_w, w1t, EMBED, HID);
    k_transpose_f16<<<dim3(PRED / 32, HID / 32), dim3(32, 8), 0, stream>>>(
        fc2_w, w2t, HID, PRED);
    k_spectral<<<ROWS * NWIN, 64, 0, stream>>>(x, window, revin_w, revin_b,
                                               mean, stdv, bw, emb);
    k_fc1<<<dim3(ROWS / 16, HID / 128), 256, 0, stream>>>(emb, w1t, fc1_b, hbuf);
    k_fc2<<<(ROWS / 16) * (PRED / 16) / 4, 128, 0, stream>>>(hbuf, w2t, fc2_b,
                                                             revin_w, revin_b,
                                                             mean, stdv,
                                                             (float*)d_out);
}
